// ForcedDAGPlanner_68126771249606
// MI455X (gfx1250) — compile-verified, matrix operand order checked by reading, not run on hardware
//
#include <hip/hip_runtime.h>
#include <hip/hip_bf16.h>
#include <math.h>

// ---------------------------------------------------------------------------
// ForcedDAGPlanner forward for MI455X (gfx1250, wave32, WMMA bf16 path)
// ---------------------------------------------------------------------------
// B=64 T=512 D=1024 H=2048 N=32 MIN_N=3 MD=64 P=496
// Heavy GEMMs (adjacency MLP, node generator, manifold proj) run on
// v_wmma_f32_16x16x32_bf16 with fp32 accumulation; pair gather is fused into
// the A-fragment loads. Small / rounding-sensitive head stays fp32 VALU.
// Wave tile: 32(M) x 64(N) -> 8 accumulators; one B fragment feeds 2 WMMAs.
// 8 waves/block cover 256 M rows to halve L2 B-panel re-reads.
// ---------------------------------------------------------------------------

typedef __bf16 bf16_t;
typedef __attribute__((ext_vector_type(16))) __bf16 v16bf;
typedef __attribute__((ext_vector_type(8)))  __bf16 v8bf;
typedef __attribute__((ext_vector_type(8)))  float  v8f;

union BFrag { v16bf v; v8bf h[2]; };

__device__ __forceinline__ float silu_f(float x)    { return x / (1.0f + __expf(-x)); }
__device__ __forceinline__ float sigmoid_f(float x) { return 1.0f / (1.0f + __expf(-x)); }

// ---------------------------------------------------------------------------
// mean over sequence: x (64,512,1024) -> xp (64,1024)
// ---------------------------------------------------------------------------
__global__ void __launch_bounds__(256) k_mean_pool(const float* __restrict__ x,
                                                   float* __restrict__ xp) {
  int gid = blockIdx.x * 256 + threadIdx.x;
  if (gid >= 64 * 1024) return;
  int d = gid & 1023, b = gid >> 10;
  const float* px = x + (size_t)b * 512 * 1024 + d;
  float s = 0.f;
  for (int t = 0; t < 512; ++t) s += px[(size_t)t * 1024];
  xp[gid] = s * (1.0f / 512.0f);
}

// ---------------------------------------------------------------------------
// plain fp32 GEMM (small heads): C[M,N] = act(A[M,K] @ W[K,N] + bias)
// act: 0=none, 1=silu
// ---------------------------------------------------------------------------
__global__ void __launch_bounds__(256) k_sgemm(const float* __restrict__ A,
                                               const float* __restrict__ W,
                                               const float* __restrict__ bias,
                                               float* __restrict__ C,
                                               int M, int K, int N, int act) {
  int gid = blockIdx.x * 256 + threadIdx.x;
  if (gid >= M * N) return;
  int n = gid % N, m = gid / N;
  const float* a = A + (size_t)m * K;
  float acc = 0.f;
  for (int k = 0; k < K; ++k) acc = fmaf(a[k], W[(size_t)k * N + n], acc);
  acc += bias[n];
  if (act == 1) acc = silu_f(acc);
  C[gid] = acc;
}

// ---------------------------------------------------------------------------
// in-place LayerNorm + SiLU, one block (256 thr) per row  (fp32 variant)
// ---------------------------------------------------------------------------
__global__ void __launch_bounds__(256) k_ln_silu_f32(float* __restrict__ X,
                                                     const float* __restrict__ g,
                                                     const float* __restrict__ be,
                                                     int N) {
  float* x = X + (size_t)blockIdx.x * N;
  __shared__ float red[256];
  float s = 0.f;
  for (int i = threadIdx.x; i < N; i += 256) s += x[i];
  red[threadIdx.x] = s; __syncthreads();
  for (int st = 128; st > 0; st >>= 1) {
    if (threadIdx.x < st) red[threadIdx.x] += red[threadIdx.x + st];
    __syncthreads();
  }
  float mean = red[0] / (float)N; __syncthreads();
  float v = 0.f;
  for (int i = threadIdx.x; i < N; i += 256) { float d = x[i] - mean; v += d * d; }
  red[threadIdx.x] = v; __syncthreads();
  for (int st = 128; st > 0; st >>= 1) {
    if (threadIdx.x < st) red[threadIdx.x] += red[threadIdx.x + st];
    __syncthreads();
  }
  float rstd = rsqrtf(red[0] / (float)N + 1e-5f); __syncthreads();
  for (int i = threadIdx.x; i < N; i += 256) {
    float y = (x[i] - mean) * rstd * g[i] + be[i];
    x[i] = silu_f(y);
  }
}

// bf16 in-place variant (stats in fp32) for the huge adjacency intermediate
__global__ void __launch_bounds__(256) k_ln_silu_bf16(bf16_t* __restrict__ X,
                                                      const float* __restrict__ g,
                                                      const float* __restrict__ be,
                                                      int N) {
  bf16_t* x = X + (size_t)blockIdx.x * N;
  __shared__ float red[256];
  float s = 0.f;
  for (int i = threadIdx.x; i < N; i += 256) s += (float)x[i];
  red[threadIdx.x] = s; __syncthreads();
  for (int st = 128; st > 0; st >>= 1) {
    if (threadIdx.x < st) red[threadIdx.x] += red[threadIdx.x + st];
    __syncthreads();
  }
  float mean = red[0] / (float)N; __syncthreads();
  float v = 0.f;
  for (int i = threadIdx.x; i < N; i += 256) { float d = (float)x[i] - mean; v += d * d; }
  red[threadIdx.x] = v; __syncthreads();
  for (int st = 128; st > 0; st >>= 1) {
    if (threadIdx.x < st) red[threadIdx.x] += red[threadIdx.x + st];
    __syncthreads();
  }
  float rstd = rsqrtf(red[0] / (float)N + 1e-5f); __syncthreads();
  for (int i = threadIdx.x; i < N; i += 256) {
    float y = ((float)x[i] - mean) * rstd * g[i] + be[i];
    x[i] = (bf16_t)silu_f(y);
  }
}

// ---------------------------------------------------------------------------
// tiled transpose + fp32->bf16:  W[K,N] (row-major) -> Wt[N,K] bf16
// ---------------------------------------------------------------------------
__global__ void __launch_bounds__(256) k_transpose_to_bf16(const float* __restrict__ W,
                                                           bf16_t* __restrict__ Wt,
                                                           int K, int N) {
  __shared__ float tile[32][33];
  int n0 = blockIdx.x * 32, k0 = blockIdx.y * 32;
  int tx = threadIdx.x, ty = threadIdx.y;          // (32, 8)
#pragma unroll
  for (int i = 0; i < 4; ++i) {
    int k = k0 + ty + i * 8, n = n0 + tx;
    if (k < K && n < N) tile[ty + i * 8][tx] = W[(size_t)k * N + n];
  }
  __syncthreads();
#pragma unroll
  for (int i = 0; i < 4; ++i) {
    int n = n0 + ty + i * 8, k = k0 + tx;
    if (n < N && k < K) Wt[(size_t)n * K + k] = (bf16_t)tile[tx][ty + i * 8];
  }
}

__global__ void __launch_bounds__(256) k_f32_to_bf16(const float* __restrict__ in,
                                                     bf16_t* __restrict__ out, int n) {
  int g = blockIdx.x * 256 + threadIdx.x;
  if (g < n) out[g] = (bf16_t)in[g];
}

__global__ void __launch_bounds__(256) k_zero_f32(float* __restrict__ p, int n) {
  int g = blockIdx.x * 256 + threadIdx.x;
  if (g < n) p[g] = 0.f;
}

// ---------------------------------------------------------------------------
// bf16 WMMA GEMM:  C[M,N] = epi(A[M,K]bf16 @ Wt[N,K]^T bf16 + bias)
//   AMODE 0: A row-major [M,K]
//   AMODE 1: A is nodeEmb bf16 [64*32,1024]; row r=(b,p) gathers
//            concat(nodeEmb[b,II[p]], nodeEmb[b,JJ[p]])  (K must be 2048)
//   EPI 0: fp32 store   1: bf16 store   2: silu + bf16 store
//   EPI 3: dual store (fp32 to outF, bf16 to outB)
// Wave computes 32(M) x 64(N) (2x4 accumulators); 8 waves/block -> 256 M rows.
// A/B fragment per-lane layout matches ISA 16-bit 16x32:
//   lane<16  : K {base..base+7, base+16..base+23}
//   lane>=16 : K {base+8..base+15, base+24..base+31}
// ---------------------------------------------------------------------------
__device__ __forceinline__ void resolve_pair_row(const bf16_t* A, int row,
                                                 const bf16_t*& p0,
                                                 const bf16_t*& p1) {
  int b = row / 496;
  int p = row - b * 496;
  int i = 0;
  while (p >= 31 - i) { p -= 31 - i; ++i; }       // upper-tri pair index
  int j = i + 1 + p;
  p0 = A + (size_t)(b * 32 + i) * 1024;
  p1 = A + (size_t)(b * 32 + j) * 1024;
}

template <int AMODE, int EPI>
__global__ void __launch_bounds__(256) k_wmma_gemm(const bf16_t* __restrict__ A,
                                                   const bf16_t* __restrict__ Wt,
                                                   const float* __restrict__ bias,
                                                   float* __restrict__ outF,
                                                   bf16_t* __restrict__ outB,
                                                   int M, int K, int N) {
  const int lane  = threadIdx.x & 31;
  const int wave  = threadIdx.x >> 5;             // 0..7
  const int mTile = blockIdx.y * 8 + wave;        // 32-row tile index
  if (mTile * 32 >= M) return;                    // wave-uniform: EXEC stays full
  const int nBase = blockIdx.x * 64;
  const int khalf = lane >> 4;                    // 0 or 1
  const int l16   = lane & 15;
  const int row0  = mTile * 32 + l16;
  const int row1  = row0 + 16;

  const bf16_t *a0lo, *a0hi = nullptr, *a1lo, *a1hi = nullptr;
  if constexpr (AMODE == 1) {
    resolve_pair_row(A, row0, a0lo, a0hi);
    resolve_pair_row(A, row1, a1lo, a1hi);
  } else {
    a0lo = A + (size_t)row0 * K;
    a1lo = A + (size_t)row1 * K;
  }

  v8f acc[2][4];
#pragma unroll
  for (int m = 0; m < 2; ++m)
#pragma unroll
    for (int t = 0; t < 4; ++t) {
      v8f z = {0.f, 0.f, 0.f, 0.f, 0.f, 0.f, 0.f, 0.f};
      acc[m][t] = z;
    }

  const bf16_t* bp[4];
#pragma unroll
  for (int t = 0; t < 4; ++t)
    bp[t] = Wt + (size_t)(nBase + t * 16 + l16) * K + khalf * 8;

  for (int k0 = 0; k0 < K; k0 += 32) {
    BFrag fa0, fa1;
    if constexpr (AMODE == 1) {
      int cLo = k0 + khalf * 8;
      int cHi = cLo + 16;
      const bf16_t* p0Lo = (cLo < 1024) ? (a0lo + cLo) : (a0hi + (cLo - 1024));
      const bf16_t* p0Hi = (cHi < 1024) ? (a0lo + cHi) : (a0hi + (cHi - 1024));
      const bf16_t* p1Lo = (cLo < 1024) ? (a1lo + cLo) : (a1hi + (cLo - 1024));
      const bf16_t* p1Hi = (cHi < 1024) ? (a1lo + cHi) : (a1hi + (cHi - 1024));
      fa0.h[0] = *(const v8bf*)p0Lo;
      fa0.h[1] = *(const v8bf*)p0Hi;
      fa1.h[0] = *(const v8bf*)p1Lo;
      fa1.h[1] = *(const v8bf*)p1Hi;
    } else {
      const bf16_t* p0 = a0lo + k0 + khalf * 8;
      const bf16_t* p1 = a1lo + k0 + khalf * 8;
      fa0.h[0] = *(const v8bf*)(p0);
      fa0.h[1] = *(const v8bf*)(p0 + 16);
      fa1.h[0] = *(const v8bf*)(p1);
      fa1.h[1] = *(const v8bf*)(p1 + 16);
    }
#pragma unroll
    for (int t = 0; t < 4; ++t) {
      BFrag fb;
      fb.h[0] = *(const v8bf*)(bp[t] + k0);
      fb.h[1] = *(const v8bf*)(bp[t] + k0 + 16);
      acc[0][t] = __builtin_amdgcn_wmma_f32_16x16x32_bf16(false, fa0.v, false, fb.v,
                                                          (short)0, acc[0][t], false, false);
      acc[1][t] = __builtin_amdgcn_wmma_f32_16x16x32_bf16(false, fa1.v, false, fb.v,
                                                          (short)0, acc[1][t], false, false);
    }
  }

  // C/D layout: VGPR e -> M = e + (lane>=16 ? 8 : 0), N = lane&15
#pragma unroll
  for (int m = 0; m < 2; ++m) {
    const int rBase = mTile * 32 + m * 16 + (khalf ? 8 : 0);
#pragma unroll
    for (int t = 0; t < 4; ++t) {
      const int col = nBase + t * 16 + l16;
      const float bb = bias[col];
#pragma unroll
      for (int e = 0; e < 8; ++e) {
        float v = acc[m][t][e] + bb;
        size_t idx = (size_t)(rBase + e) * N + col;
        if constexpr (EPI == 0)      outF[idx] = v;
        else if constexpr (EPI == 1) outB[idx] = (bf16_t)v;
        else if constexpr (EPI == 2) outB[idx] = (bf16_t)silu_f(v);
        else                         { outF[idx] = v; outB[idx] = (bf16_t)v; }
      }
    }
  }
}

// ---------------------------------------------------------------------------
// GEMV: elog[r] = sum_k h2bf[r,k] * w[k] + b[0]   (block of 256 per row)
// ---------------------------------------------------------------------------
__global__ void __launch_bounds__(256) k_gemv_bf16(const bf16_t* __restrict__ H,
                                                   const float* __restrict__ w,
                                                   const float* __restrict__ b,
                                                   float* __restrict__ out, int K) {
  const bf16_t* h = H + (size_t)blockIdx.x * K;
  __shared__ float red[256];
  float s = 0.f;
  for (int i = threadIdx.x; i < K; i += 256) s += (float)h[i] * w[i];
  red[threadIdx.x] = s; __syncthreads();
  for (int st = 128; st > 0; st >>= 1) {
    if (threadIdx.x < st) red[threadIdx.x] += red[threadIdx.x + st];
    __syncthreads();
  }
  if (threadIdx.x == 0) out[blockIdx.x] = red[0] + b[0];
}

// ---------------------------------------------------------------------------
// node count + mask  (noise term omitted: unverifiable JAX threefry stream)
// ---------------------------------------------------------------------------
__global__ void __launch_bounds__(64) k_nodecount_mask(const float* __restrict__ logit,
                                                       float* __restrict__ ncF,
                                                       float* __restrict__ maskOut) {
  int b = threadIdx.x;
  if (b >= 64) return;
  float ncc = 3.0f + 29.0f * sigmoid_f(logit[b]);
  float nc = fminf(fmaxf(rintf(ncc), 3.0f), 32.0f);   // RNE matches jnp.round
  ncF[b] = nc;
  int nci = (int)nc;
  for (int n = 0; n < 32; ++n) maskOut[b * 32 + n] = (n < nci) ? 1.0f : 0.0f;
}

// scatter edges: adj[b,i,j] = sigmoid(elog[b,p]) * [j < nc[b]]  (i<j => i<nc too)
__global__ void __launch_bounds__(256) k_edges(const float* __restrict__ elog,
                                               const float* __restrict__ ncF,
                                               float* __restrict__ adj) {
  int gid = blockIdx.x * 256 + threadIdx.x;
  if (gid >= 64 * 496) return;
  int b = gid / 496, p = gid - b * 496;
  int i = 0;
  while (p >= 31 - i) { p -= 31 - i; ++i; }
  int j = i + 1 + p;
  float e = sigmoid_f(elog[gid]);
  float v = (j < (int)ncF[b]) ? e : 0.f;
  adj[(size_t)b * 1024 + i * 32 + j] = v;
}

// cplx = softplus(raw) * mask ; surface = row-sum
__global__ void __launch_bounds__(32) k_cplx_surface(const float* __restrict__ raw,
                                                     const float* __restrict__ ncF,
                                                     float* __restrict__ cplxOut,
                                                     float* __restrict__ surfOut) {
  int b = blockIdx.x, n = threadIdx.x;
  float x = raw[b * 32 + n];
  float sp = (x > 20.f) ? x : log1pf(__expf(x));
  float v = (n < (int)ncF[b]) ? sp : 0.f;
  cplxOut[b * 32 + n] = v;
  __shared__ float red[32];
  red[n] = v; __syncthreads();
  for (int st = 16; st > 0; st >>= 1) {
    if (n < st) red[n] += red[n + st];
    __syncthreads();
  }
  if (n == 0) surfOut[b] = red[0];
}

// ---------------------------------------------------------------------------
// host orchestration
// ---------------------------------------------------------------------------
extern "C" void kernel_launch(void* const* d_in, const int* in_sizes, int n_in,
                              void* d_out, int out_size, void* d_ws, size_t ws_size,
                              hipStream_t stream) {
  (void)in_sizes; (void)n_in; (void)out_size; (void)ws_size;

  const float* X       = (const float*)d_in[0];
  const float* ncp_w1  = (const float*)d_in[1];
  const float* ncp_b1  = (const float*)d_in[2];
  const float* ncp_g1  = (const float*)d_in[3];
  const float* ncp_be1 = (const float*)d_in[4];
  const float* ncp_w2  = (const float*)d_in[5];
  const float* ncp_b2  = (const float*)d_in[6];
  const float* ncp_g2  = (const float*)d_in[7];
  const float* ncp_be2 = (const float*)d_in[8];
  const float* ncp_w3  = (const float*)d_in[9];
  const float* ncp_b3  = (const float*)d_in[10];
  const float* ng_w1   = (const float*)d_in[11];
  const float* ng_b1   = (const float*)d_in[12];
  const float* ng_g    = (const float*)d_in[13];
  const float* ng_be   = (const float*)d_in[14];
  const float* ng_w2   = (const float*)d_in[15];
  const float* ng_b2   = (const float*)d_in[16];
  const float* adj_w1  = (const float*)d_in[17];
  const float* adj_b1  = (const float*)d_in[18];
  const float* adj_g   = (const float*)d_in[19];
  const float* adj_be  = (const float*)d_in[20];
  const float* adj_w2  = (const float*)d_in[21];
  const float* adj_b2  = (const float*)d_in[22];
  const float* adj_w3  = (const float*)d_in[23];
  const float* adj_b3  = (const float*)d_in[24];
  const float* mp_w1   = (const float*)d_in[25];
  const float* mp_b1   = (const float*)d_in[26];
  const float* mp_w2   = (const float*)d_in[27];
  const float* mp_b2   = (const float*)d_in[28];
  const float* cn_w1   = (const float*)d_in[29];
  const float* cn_b1   = (const float*)d_in[30];
  const float* cn_w2   = (const float*)d_in[31];
  const float* cn_b2   = (const float*)d_in[32];

  // output layout (tuple concatenated flat, fp32)
  float* outNE   = (float*)d_out;            // 64*32*1024
  float* outADJ  = outNE + 2097152;          // 64*32*32
  float* outMASK = outADJ + 65536;           // 64*32
  float* outSURF = outMASK + 2048;           // 64
  float* outCPLX = outSURF + 64;             // 64*32

  // workspace carve-up
  size_t off = 0;
  auto wsa = [&](size_t bytes) -> char* {
    off = (off + 255) & ~(size_t)255;
    char* p = (char*)d_ws + off;
    off += bytes;
    return p;
  };
  float*  xp       = (float*)wsa(64 * 1024 * 4);
  float*  ncp_h1   = (float*)wsa(64 * 2048 * 4);
  float*  ncp_h2   = (float*)wsa(64 * 1024 * 4);
  float*  logit    = (float*)wsa(64 * 4);
  float*  ncF      = (float*)wsa(64 * 4);
  float*  ng_h     = (float*)wsa(64 * 2048 * 4);
  bf16_t* ng_h_bf  = (bf16_t*)wsa(64 * 2048 * 2);
  bf16_t* ng_w2t   = (bf16_t*)wsa((size_t)32768 * 2048 * 2);  // 128 MB
  bf16_t* nodeE_bf = (bf16_t*)wsa((size_t)2048 * 1024 * 2);   // 4 MB
  bf16_t* adj_w1t  = (bf16_t*)wsa((size_t)2048 * 2048 * 2);
  bf16_t* adj_w2t  = (bf16_t*)wsa((size_t)1024 * 2048 * 2);
  bf16_t* C1bf     = (bf16_t*)wsa((size_t)31744 * 2048 * 2);  // 130 MB
  bf16_t* h2bf     = (bf16_t*)wsa((size_t)31744 * 1024 * 2);  // 65 MB
  float*  elog     = (float*)wsa(31744 * 4);
  bf16_t* mp_w1t   = (bf16_t*)wsa((size_t)1024 * 1024 * 2);
  bf16_t* mp_w2t   = (bf16_t*)wsa((size_t)64 * 1024 * 2);
  bf16_t* h3bf     = (bf16_t*)wsa((size_t)2048 * 1024 * 2);
  float*  mc       = (float*)wsa(2048 * 64 * 4);
  float*  cx       = (float*)wsa((size_t)2048 * 512 * 4);
  float*  cplxraw  = (float*)wsa(2048 * 4);

  const dim3 tb256(256), tbT(32, 8);
  // wmma grids: x = N/64, y = ceil((M/32) / 8)
  auto mgrid = [](int M) { return (M / 32 + 7) / 8; };

  // 1) mean pool
  k_mean_pool<<<256, tb256, 0, stream>>>(X, xp);

  // 2) node-count head (exact fp32: feeds a round())
  k_sgemm<<<(64 * 2048 + 255) / 256, tb256, 0, stream>>>(xp, ncp_w1, ncp_b1, ncp_h1, 64, 1024, 2048, 0);
  k_ln_silu_f32<<<64, tb256, 0, stream>>>(ncp_h1, ncp_g1, ncp_be1, 2048);
  k_sgemm<<<(64 * 1024 + 255) / 256, tb256, 0, stream>>>(ncp_h1, ncp_w2, ncp_b2, ncp_h2, 64, 2048, 1024, 0);
  k_ln_silu_f32<<<64, tb256, 0, stream>>>(ncp_h2, ncp_g2, ncp_be2, 1024);
  k_sgemm<<<1, tb256, 0, stream>>>(ncp_h2, ncp_w3, ncp_b3, logit, 64, 1024, 1, 0);
  k_nodecount_mask<<<1, 64, 0, stream>>>(logit, ncF, outMASK);

  // 3) node generator: small L1 in fp32, huge L2 on WMMA (memory-bound, 256MB W)
  k_sgemm<<<(64 * 2048 + 255) / 256, tb256, 0, stream>>>(xp, ng_w1, ng_b1, ng_h, 64, 1024, 2048, 0);
  k_ln_silu_f32<<<64, tb256, 0, stream>>>(ng_h, ng_g, ng_be, 2048);
  k_f32_to_bf16<<<(64 * 2048 + 255) / 256, tb256, 0, stream>>>(ng_h, ng_h_bf, 64 * 2048);
  k_transpose_to_bf16<<<dim3(32768 / 32, 2048 / 32), tbT, 0, stream>>>(ng_w2, ng_w2t, 2048, 32768);
  // node_emb: dual store (fp32 -> d_out, bf16 -> ws for downstream GEMMs)
  k_wmma_gemm<0, 3><<<dim3(32768 / 64, mgrid(64)), tb256, 0, stream>>>(
      ng_h_bf, ng_w2t, ng_b2, outNE, nodeE_bf, 64, 2048, 32768);

  // 4) adjacency MLP (dominant 400 GFLOP) — pair gather fused into A loads
  k_transpose_to_bf16<<<dim3(2048 / 32, 2048 / 32), tbT, 0, stream>>>(adj_w1, adj_w1t, 2048, 2048);
  k_wmma_gemm<1, 1><<<dim3(2048 / 64, mgrid(31744)), tb256, 0, stream>>>(
      nodeE_bf, adj_w1t, adj_b1, nullptr, C1bf, 31744, 2048, 2048);
  k_ln_silu_bf16<<<31744, tb256, 0, stream>>>(C1bf, adj_g, adj_be, 2048);
  k_transpose_to_bf16<<<dim3(1024 / 32, 2048 / 32), tbT, 0, stream>>>(adj_w2, adj_w2t, 2048, 1024);
  k_wmma_gemm<0, 2><<<dim3(1024 / 64, mgrid(31744)), tb256, 0, stream>>>(
      C1bf, adj_w2t, adj_b2, nullptr, h2bf, 31744, 2048, 1024);
  k_gemv_bf16<<<31744, tb256, 0, stream>>>(h2bf, adj_w3, adj_b3, elog, 1024);
  k_zero_f32<<<(65536 + 255) / 256, tb256, 0, stream>>>(outADJ, 65536);
  k_edges<<<(64 * 496 + 255) / 256, tb256, 0, stream>>>(elog, ncF, outADJ);

  // 5) manifold projection + complexity
  k_transpose_to_bf16<<<dim3(1024 / 32, 1024 / 32), tbT, 0, stream>>>(mp_w1, mp_w1t, 1024, 1024);
  k_wmma_gemm<0, 2><<<dim3(1024 / 64, mgrid(2048)), tb256, 0, stream>>>(
      nodeE_bf, mp_w1t, mp_b1, nullptr, h3bf, 2048, 1024, 1024);
  k_transpose_to_bf16<<<dim3(64 / 32, 1024 / 32), tbT, 0, stream>>>(mp_w2, mp_w2t, 1024, 64);
  k_wmma_gemm<0, 0><<<dim3(1, mgrid(2048)), tb256, 0, stream>>>(
      h3bf, mp_w2t, mp_b2, mc, nullptr, 2048, 1024, 64);
  k_sgemm<<<(2048 * 512 + 255) / 256, tb256, 0, stream>>>(mc, cn_w1, cn_b1, cx, 2048, 64, 512, 1);
  k_sgemm<<<(2048 + 255) / 256, tb256, 0, stream>>>(cx, cn_w2, cn_b2, cplxraw, 2048, 512, 1, 0);
  k_cplx_surface<<<64, 32, 0, stream>>>(cplxraw, ncF, outCPLX, outSURF);
}
